// MyBilinear_24790551232601
// MI455X (gfx1250) — matrix operand order repaired and checked by reading, not compiled, not run on hardware
//
#include <hip/hip_runtime.h>

#define B_   8
#define L_   256
#define P_   512
#define Q_   512
#define K_   64

typedef __attribute__((ext_vector_type(16))) __bf16 v16bf;
typedef __attribute__((ext_vector_type(8)))  float  v8f;
typedef __attribute__((ext_vector_type(4)))  unsigned int uint4v;

__device__ __forceinline__ unsigned short f32_to_bf16(float f) {
  unsigned int u = __float_as_uint(f);
  unsigned int r = u + 0x7FFFu + ((u >> 16) & 1u);   // round-to-nearest-even
  return (unsigned short)(r >> 16);
}

__device__ __forceinline__ unsigned int pack_bf16x2(float lo, float hi) {
  return (unsigned int)f32_to_bf16(lo) | ((unsigned int)f32_to_bf16(hi) << 16);
}

// A-fragment (16-bit A 16x32 layout): K(e) = e + (e>=8?8:0) + lanehi8.
// Two contiguous 8-element (16B) runs at +0 and +16 elements.
__device__ __forceinline__ v16bf load_frag_split(const unsigned short* p) {
  union { v16bf v; uint4v u[2]; } r;
  r.u[0] = *(const uint4v*)(p);
  r.u[1] = *(const uint4v*)(p + 16);
  return r.v;
}

// B-fragment (16-bit B 32x16 layout): K(e) = e + (e>=8?8:0) + lanehi8.
// Two contiguous 8-element (16B) runs at +0 and +16 elements.
__device__ __forceinline__ v16bf load_frag_contig(const unsigned short* p) {
  union { v16bf v; uint4v u[2]; } r;
  r.u[0] = *(const uint4v*)(p);
  r.u[1] = *(const uint4v*)(p + 16);
  return r.v;
}

// ---------------------------------------------------------------------------
// Precompute kernels (run once per launch; all reads/writes coalesced)
// ---------------------------------------------------------------------------

// wsX1[b][i][p] = bf16(x1[b][i][p])   (linear copy, 1M elements)
__global__ __launch_bounds__(256)
void convert_x1_bf16(const float* __restrict__ src, unsigned short* __restrict__ dst) {
  size_t idx = ((size_t)blockIdx.x * 256 + threadIdx.x) * 8;
  const float4* s4 = (const float4*)(src + idx);
  float4 a = s4[0], b = s4[1];
  uint4v o;
  o.x = pack_bf16x2(a.x, a.y);
  o.y = pack_bf16x2(a.z, a.w);
  o.z = pack_bf16x2(b.x, b.y);
  o.w = pack_bf16x2(b.z, b.w);
  *(uint4v*)(dst + idx) = o;
}

// wsX2[((b*16+qb)*256 + j)*32 + ql] = bf16(x2[b][j][qb*32+ql])
__global__ __launch_bounds__(256)
void convert_x2_blocked(const float* __restrict__ x2, unsigned short* __restrict__ dst) {
  int blk = blockIdx.x;              // b*16 + qb
  int b = blk >> 4, qb = blk & 15;
  int j = threadIdx.x;               // 0..255
  const float4* s4 = (const float4*)(x2 + ((size_t)b * L_ + j) * Q_ + qb * 32);
  unsigned int row[16];
#pragma unroll
  for (int t = 0; t < 8; ++t) {
    float4 v = s4[t];
    row[2 * t + 0] = pack_bf16x2(v.x, v.y);
    row[2 * t + 1] = pack_bf16x2(v.z, v.w);
  }
  uint4v* d4 = (uint4v*)(dst + ((size_t)blk * 256 + j) * 32);
#pragma unroll
  for (int t = 0; t < 4; ++t) {
    uint4v o;
    o.x = row[4 * t + 0]; o.y = row[4 * t + 1];
    o.z = row[4 * t + 2]; o.w = row[4 * t + 3];
    d4[t] = o;
  }
}

// wsW[((k*16+pb)*512 + q)*32 + pl] = bf16(W[k][pb*32+pl][q])
// (32 p x 512 q block transpose through padded LDS tile)
__global__ __launch_bounds__(256)
void convert_W_frag(const float* __restrict__ W, unsigned short* __restrict__ dst) {
  __shared__ unsigned short tile[32 * 514];   // pitch 514 avoids bank conflicts
  const int k   = blockIdx.x;   // 0..63
  const int pb  = blockIdx.y;   // 0..15
  const int tid = threadIdx.x;
  {
    int pl = tid >> 3;                 // 0..31
    int q0 = (tid & 7) * 64;
    const float4* s4 = (const float4*)(W + ((size_t)k * P_ + pb * 32 + pl) * Q_ + q0);
    unsigned short* trow = &tile[pl * 514 + q0];
#pragma unroll
    for (int t = 0; t < 16; ++t) {
      float4 v = s4[t];
      trow[4 * t + 0] = f32_to_bf16(v.x);
      trow[4 * t + 1] = f32_to_bf16(v.y);
      trow[4 * t + 2] = f32_to_bf16(v.z);
      trow[4 * t + 3] = f32_to_bf16(v.w);
    }
  }
  __syncthreads();
#pragma unroll
  for (int qq = 0; qq < 2; ++qq) {
    int q = tid * 2 + qq;              // 0..511
    unsigned int row[16];
#pragma unroll
    for (int t = 0; t < 16; ++t) {
      unsigned int lo = tile[(2 * t + 0) * 514 + q];
      unsigned int hi = tile[(2 * t + 1) * 514 + q];
      row[t] = lo | (hi << 16);
    }
    uint4v* d4 = (uint4v*)(dst + (((size_t)(k * 16 + pb)) * 512 + q) * 32);
#pragma unroll
    for (int t = 0; t < 4; ++t) {
      uint4v o;
      o.x = row[4 * t + 0]; o.y = row[4 * t + 1];
      o.z = row[4 * t + 2]; o.w = row[4 * t + 3];
      d4[t] = o;
    }
  }
}

// ---------------------------------------------------------------------------
// Fast fused kernel: all operands pre-converted bf16 in fragment-native layout.
// Two i-tiles (32 rows) per workgroup: every B fragment feeds two WMMAs.
// Stage 1 is barrier-free; LDS only holds the shared 32x512 T tile.
// ---------------------------------------------------------------------------
__global__ __launch_bounds__(256)
void bilinear_fused_fast(const unsigned short* __restrict__ x1b,  // [b][i][p]
                         const unsigned short* __restrict__ x2b,  // [b][qb][j][ql]
                         const unsigned short* __restrict__ wb,   // [k][pb][q][pl]
                         const float* __restrict__ bias,
                         float* __restrict__ out) {
  __shared__ unsigned short sT[32 * 512];   // 32 KB: T tile [32 i][512 q] bf16

  const int tid     = threadIdx.x;
  const int lane    = tid & 31;
  const int wave    = tid >> 5;              // 0..7
  const int lanehi8 = (lane < 16) ? 0 : 8;   // A-frag K offset AND C/D row offset
  const int kbase16 = (lane < 16) ? 0 : 16;  // unused here (B-frag K offset is lanehi8)
  const int lcol    = lane & 15;

  const int itile = blockIdx.x;   // 0..7  (pair of 16-row i-tiles)
  const int k     = blockIdx.y;   // 0..63
  const int b     = blockIdx.z;   // 0..7

  const v8f vzero = {0.f, 0.f, 0.f, 0.f, 0.f, 0.f, 0.f, 0.f};

  // ---- stage 1: T = X1tile (32xP) * W_k (PxQ) ----
  v8f acc0[4], acc1[4];
#pragma unroll
  for (int t = 0; t < 4; ++t) { acc0[t] = vzero; acc1[t] = vzero; }

  const unsigned short* arow0 =
      x1b + ((size_t)b * L_ + (size_t)itile * 32 + lcol) * P_;
  const unsigned short* arow1 = arow0 + (size_t)16 * P_;
  const unsigned short* wk = wb + (size_t)k * (16 * 512 * 32);

  for (int pb = 0; pb < 16; ++pb) {
    const unsigned short* wblk = wk + (size_t)pb * (512 * 32);
    if (pb + 1 < 16) {  // prefetch next 32 KB W slab into near caches (WGP scope)
      const unsigned short* nxt = wblk + 512 * 32 + (size_t)tid * 64;
      __builtin_prefetch(nxt, 0, 3);
      __builtin_prefetch(nxt + 32, 0, 3);
    }
    v16bf a0 = load_frag_split(arow0 + pb * 32 + lanehi8);
    v16bf a1 = load_frag_split(arow1 + pb * 32 + lanehi8);
#pragma unroll
    for (int t = 0; t < 4; ++t) {
      int q = (wave * 4 + t) * 16 + lcol;
      v16bf bm = load_frag_contig(wblk + q * 32 + lanehi8);
      acc0[t] = __builtin_amdgcn_wmma_f32_16x16x32_bf16(
          false, a0, false, bm, (short)0, acc0[t], false, false);
      acc1[t] = __builtin_amdgcn_wmma_f32_16x16x32_bf16(
          false, a1, false, bm, (short)0, acc1[t], false, false);
    }
  }

  // ---- spill T (bf16) to LDS for cross-wave sharing ----
#pragma unroll
  for (int t = 0; t < 4; ++t) {
    int n = (wave * 4 + t) * 16 + lcol;
#pragma unroll
    for (int r = 0; r < 8; ++r) {
      sT[(lanehi8 + r) * 512 + n]      = f32_to_bf16(acc0[t][r]);
      sT[(16 + lanehi8 + r) * 512 + n] = f32_to_bf16(acc1[t][r]);
    }
  }
  __syncthreads();

  // ---- stage 2: out(32 i, 256 j, k) = T (32xQ) * X2_b^T (Qx256) ----
  v8f o00 = vzero, o01 = vzero, o10 = vzero, o11 = vzero;
  const unsigned short* x2base = x2b + (size_t)b * (16 * 256 * 32);
  for (int qb = 0; qb < 16; ++qb) {
    const unsigned short* xblk = x2base + (size_t)qb * (256 * 32);
    v16bf a0 = load_frag_split(&sT[lcol * 512 + qb * 32 + lanehi8]);
    v16bf a1 = load_frag_split(&sT[(16 + lcol) * 512 + qb * 32 + lanehi8]);
    {
      int j = (wave * 2 + 0) * 16 + lcol;
      v16bf bm = load_frag_contig(xblk + j * 32 + lanehi8);
      o00 = __builtin_amdgcn_wmma_f32_16x16x32_bf16(
          false, a0, false, bm, (short)0, o00, false, false);
      o10 = __builtin_amdgcn_wmma_f32_16x16x32_bf16(
          false, a1, false, bm, (short)0, o10, false, false);
    }
    {
      int j = (wave * 2 + 1) * 16 + lcol;
      v16bf bm = load_frag_contig(xblk + j * 32 + lanehi8);
      o01 = __builtin_amdgcn_wmma_f32_16x16x32_bf16(
          false, a0, false, bm, (short)0, o01, false, false);
      o11 = __builtin_amdgcn_wmma_f32_16x16x32_bf16(
          false, a1, false, bm, (short)0, o11, false, false);
    }
  }

  // ---- store out[b,i,j,k] + bias[k] (k-strided scatter absorbed by L2) ----
  const float bv = bias[k];
#pragma unroll
  for (int h = 0; h < 2; ++h) {
#pragma unroll
    for (int t = 0; t < 2; ++t) {
      v8f c = h ? (t ? o11 : o10) : (t ? o01 : o00);
      int j = (wave * 2 + t) * 16 + lcol;
      size_t base =
          (((size_t)b * L_ + (size_t)itile * 32 + h * 16 + lanehi8) * L_ + j) * K_ + k;
#pragma unroll
      for (int r = 0; r < 8; ++r) {
        out[base + (size_t)r * (L_ * K_)] = c[r] + bv;
      }
    }
  }
}

// ---------------------------------------------------------------------------
// Fallback fused kernel (no workspace needed): converts fp32->bf16 on the fly.
// ---------------------------------------------------------------------------
__device__ __forceinline__ v16bf load_b_gather(const unsigned short* s, int ldn,
                                               int col, int kbase16) {
  union { v16bf v; unsigned short u[16]; } r;
#pragma unroll
  for (int e = 0; e < 16; ++e) r.u[e] = s[(e + (e >= 8 ? 8 : 0) + kbase16 / 2) * ldn + col];
  return r.v;
}

__global__ __launch_bounds__(256)
void bilinear_fused_direct(const float* __restrict__ x1,
                           const float* __restrict__ x2,
                           const float* __restrict__ W,
                           const float* __restrict__ bias,
                           float* __restrict__ out) {
  __shared__ unsigned short sX1[16 * 512];
  __shared__ unsigned short sB[32 * 512];

  const int tid     = threadIdx.x;
  const int lane    = tid & 31;
  const int wave    = tid >> 5;
  const int lanehi8 = (lane < 16) ? 0 : 8;
  const int kbase16 = (lane < 16) ? 0 : 16;
  const int lcol    = lane & 15;

  const int itile = blockIdx.x;
  const int k     = blockIdx.y;
  const int b     = blockIdx.z;

  {
    int row = tid >> 4;
    int c0  = (tid & 15) * 32;
    const float4* s4 = (const float4*)(x1 + ((size_t)b * L_ + (size_t)itile * 16 + row) * P_ + c0);
    unsigned short* dst = &sX1[row * 512 + c0];
#pragma unroll
    for (int t = 0; t < 8; ++t) {
      float4 v = s4[t];
      dst[t * 4 + 0] = f32_to_bf16(v.x);
      dst[t * 4 + 1] = f32_to_bf16(v.y);
      dst[t * 4 + 2] = f32_to_bf16(v.z);
      dst[t * 4 + 3] = f32_to_bf16(v.w);
    }
  }

  const float* Wk = W + (size_t)k * P_ * Q_;
  const v8f vzero = {0.f, 0.f, 0.f, 0.f, 0.f, 0.f, 0.f, 0.f};
  v8f acc[4];
#pragma unroll
  for (int t = 0; t < 4; ++t) acc[t] = vzero;

  for (int kp = 0; kp < P_ / 32; ++kp) {
    const int p0 = kp * 32;
    {
      int row = tid >> 3;
      int c0  = (tid & 7) * 64;
      const float4* w4 = (const float4*)(Wk + (size_t)(p0 + row) * Q_ + c0);
      unsigned short* dst = &sB[row * 512 + c0];
#pragma unroll
      for (int t = 0; t < 16; ++t) {
        float4 v = w4[t];
        dst[t * 4 + 0] = f32_to_bf16(v.x);
        dst[t * 4 + 1] = f32_to_bf16(v.y);
        dst[t * 4 + 2] = f32_to_bf16(v.z);
        dst[t * 4 + 3] = f32_to_bf16(v.w);
      }
    }
    __syncthreads();
    v16bf a = load_frag_split(&sX1[lcol * 512 + p0 + lanehi8]);
#pragma unroll
    for (int t = 0; t < 4; ++t) {
      v16bf bm = load_b_gather(sB, 512, (wave * 4 + t) * 16 + lcol, kbase16);
      acc[t] = __builtin_amdgcn_wmma_f32_16x16x32_bf16(
          false, a, false, bm, (short)0, acc[t], false, false);
    }
    __syncthreads();
  }

  unsigned short* sT  = sB;
  unsigned short* sX2 = sB + 16 * 512;
#pragma unroll
  for (int t = 0; t < 4; ++t) {
    int n = (wave * 4 + t) * 16 + lcol;
#pragma unroll
    for (int r = 0; r < 8; ++r)
      sT[(lanehi8 + r) * 512 + n] = f32_to_bf16(acc[t][r]);
  }
  __syncthreads();

  v8f o0 = vzero, o1 = vzero;
  for (int kq = 0; kq < Q_ / 32; ++kq) {
    const int q0 = kq * 32;
    {
      int j = tid;
      const float4* s4 = (const float4*)(x2 + ((size_t)b * L_ + j) * Q_ + q0);
#pragma unroll
      for (int t = 0; t < 8; ++t) {
        float4 v = s4[t];
        int ql = t * 4;
        sX2[(ql + 0) * 256 + j] = f32_to_bf16(v.x);
        sX2[(ql + 1) * 256 + j] = f32_to_bf16(v.y);
        sX2[(ql + 2) * 256 + j] = f32_to_bf16(v.z);
        sX2[(ql + 3) * 256 + j] = f32_to_bf16(v.w);
      }
    }
    __syncthreads();
    v16bf a = load_frag_split(&sT[lcol * 512 + q0 + lanehi8]);
    {
      v16bf bm = load_b_gather(sX2, 256, (wave * 2 + 0) * 16 + lcol, kbase16);
      o0 = __builtin_amdgcn_wmma_f32_16x16x32_bf16(
          false, a, false, bm, (short)0, o0, false, false);
    }
    {
      v16bf bm = load_b_gather(sX2, 256, (wave * 2 + 1) * 16 + lcol, kbase16);
      o1 = __builtin_amdgcn_wmma_f32_16x16x32_bf16(
          false, a, false, bm, (short)0, o1, false, false);
    }
    __syncthreads();
  }

  const float bv = bias[k];
#pragma unroll
  for (int t = 0; t < 2; ++t) {
    v8f c = t ? o1 : o0;
    int j = (wave * 2 + t) * 16 + lcol;
    size_t base = (((size_t)b * L_ + (size_t)itile * 16 + lanehi8) * L_ + j) * K_ + k;
#pragma unroll
    for (int r = 0; r < 8; ++r) {
      out[base + (size_t)r * (L_ * K_)] = c[r] + bv;
    }
  }
}

// ---------------------------------------------------------------------------

extern "C" void kernel_launch(void* const* d_in, const int* in_sizes, int n_in,
                              void* d_out, int out_size, void* d_ws, size_t ws_size,
                              hipStream_t stream) {
  (void)in_sizes; (void)n_in; (void)out_size;
  const float* x1   = (const float*)d_in[0];
  const float* x2   = (const float*)d_in[1];
  const float* W    = (const float*)d_in[2];
  const float* bias = (const float*)d_in[3];
  float* out        = (float*)d_out;

  const size_t wsX1_off = 0;                    // 2 MB
  const size_t wsX2_off = 2u * 1024 * 1024;     // 2 MB
  const size_t wsW_off  = 4u * 1024 * 1024;     // 32 MB
  const size_t ws_need  = 36u * 1024 * 1024;

  if (d_ws != nullptr && ws_size >= ws_need) {
    unsigned short* wsX1 = (unsigned short*)((char*)d_ws + wsX1_off);
    unsigned short* wsX2 = (unsigned short*)((char*)d_ws + wsX2_off);
    unsigned short* wsW  = (unsigned short*)((char*)d_ws + wsW_off);

    convert_x1_bf16<<<512, 256, 0, stream>>>(x1, wsX1);
    convert_x2_blocked<<<128, 256, 0, stream>>>(x2, wsX2);
    dim3 gw(64, 16);
    convert_W_frag<<<gw, 256, 0, stream>>>(W, wsW);
    dim3 grid(L_ / 32, K_, B_);   // (8, 64, 8) = 4096 workgroups
    bilinear_fused_fast<<<grid, 256, 0, stream>>>(wsX1, wsX2, wsW, bias, out);
  } else {
    dim3 grid(L_ / 16, K_, B_);   // (16, 64, 8)
    bilinear_fused_direct<<<grid, 256, 0, stream>>>(x1, x2, W, bias, out);
  }
}
